// MoE_21303037788228
// MI455X (gfx1250) — matrix operand order repaired and checked by reading, not compiled, not run on hardware
//
#include <hip/hip_runtime.h>
#include <hip/hip_bf16.h>
#include <math.h>

#define TOKENS 2048
#define HIDDEN 2048
#define INTER  1408
#define EXPERTS 16
#define TOPK 2
#define NROWS (TOKENS * TOPK)   // always exactly 4096 (top-k always picks K)

typedef __attribute__((ext_vector_type(16))) __bf16 v16bf;
typedef __attribute__((ext_vector_type(8)))  __bf16 v8bf;
typedef __attribute__((ext_vector_type(8)))  float  v8f;

// A-fragment: 16x32 bf16, ISA layout. lane = m + 16*kh.
// VGPR0-3 hold K = 8*kh + [0..7]; VGPR4-7 hold K = 16 + 8*kh + [0..7].
union FragA { v8bf h[2]; v16bf v; };

__device__ __forceinline__ float fast_sigmoid(float x) {
  return __builtin_amdgcn_rcpf(1.f + __expf(-x));   // v_rcp_f32, no IEEE div expansion
}

// ---------------- router: logits -> sigmoid -> top-2 -> renorm ----------------
__global__ __launch_bounds__(32) void moe_router(const float* __restrict__ X,
                                                 const float* __restrict__ Wg,
                                                 int* __restrict__ tk_id,
                                                 float* __restrict__ tk_w) {
  const int t = blockIdx.x;
  const int l = threadIdx.x;
  __shared__ float red[32][EXPERTS];
  float acc[EXPERTS];
#pragma unroll
  for (int e = 0; e < EXPERTS; ++e) acc[e] = 0.f;
  const float* xr = X + (size_t)t * HIDDEN;
  for (int h = l; h < HIDDEN; h += 32) {
    const float xv = xr[h];
    const float* wr = Wg + (size_t)h * EXPERTS;
#pragma unroll
    for (int e = 0; e < EXPERTS; ++e) acc[e] += xv * wr[e];
  }
#pragma unroll
  for (int e = 0; e < EXPERTS; ++e) red[l][e] = acc[e];
  __syncthreads();
  if (l == 0) {
    float s[EXPERTS];
#pragma unroll
    for (int e = 0; e < EXPERTS; ++e) {
      float v = 0.f;
      for (int i = 0; i < 32; ++i) v += red[i][e];   // fixed order: deterministic
      s[e] = fast_sigmoid(v);
    }
    int i0 = 0;
    for (int e = 1; e < EXPERTS; ++e) if (s[e] > s[i0]) i0 = e;        // lowest idx wins ties
    int i1 = (i0 == 0) ? 1 : 0;
    for (int e = 0; e < EXPERTS; ++e) { if (e == i0) continue; if (s[e] > s[i1]) i1 = e; }
    const float w0 = s[i0], w1 = s[i1];
    const float inv = __builtin_amdgcn_rcpf(w0 + w1);
    tk_id[2 * t] = i0;  tk_id[2 * t + 1] = i1;
    tk_w[2 * t] = w0 * inv;  tk_w[2 * t + 1] = w1 * inv;
  }
}

// ---------------- routing metadata ----------------
__global__ void moe_zero_meta(int* __restrict__ cnt, int* __restrict__ cnt2) {
  const int i = threadIdx.x;
  if (i < EXPERTS) { cnt[i] = 0; cnt2[i] = 0; }
}

__global__ void moe_count(const int* __restrict__ tk_id, int* __restrict__ cnt) {
  const int t = blockIdx.x * blockDim.x + threadIdx.x;
  if (t >= TOKENS) return;
  atomicAdd(&cnt[tk_id[2 * t]], 1);
  atomicAdd(&cnt[tk_id[2 * t + 1]], 1);
}

__global__ void moe_prefix(const int* __restrict__ cnt, int* __restrict__ rbase) {
  if (threadIdx.x == 0 && blockIdx.x == 0) {
    int s = 0;
    for (int e = 0; e < EXPERTS; ++e) { rbase[e] = s; s += cnt[e]; }
    rbase[EXPERTS] = s;  // == NROWS
  }
}

__global__ void moe_scatter(const int* __restrict__ tk_id, const float* __restrict__ tk_w,
                            const int* __restrict__ rbase, int* __restrict__ cnt2,
                            int* __restrict__ tok_of_row, float* __restrict__ w_of_row,
                            int* __restrict__ slot_map) {
  const int t = blockIdx.x * blockDim.x + threadIdx.x;
  if (t >= TOKENS) return;
#pragma unroll
  for (int k = 0; k < TOPK; ++k) {
    const int e = tk_id[2 * t + k];
    const int slot = atomicAdd(&cnt2[e], 1);      // permutes rows only; output invariant
    const int row = rbase[e] + slot;
    tok_of_row[row] = t;
    w_of_row[row] = tk_w[2 * t + k];
    slot_map[2 * t + k] = row;
  }
}

// ---------------- X f32 -> bf16 (one pass; L2-resident afterwards) ----------------
__global__ void moe_cvt_x(const float* __restrict__ X, __bf16* __restrict__ Xb) {
  const size_t i = ((size_t)blockIdx.x * blockDim.x + threadIdx.x) * 8;
  const float4 a = *(const float4*)(X + i);
  const float4 b = *(const float4*)(X + i + 4);
  v8bf o;
  o[0] = (__bf16)a.x; o[1] = (__bf16)a.y; o[2] = (__bf16)a.z; o[3] = (__bf16)a.w;
  o[4] = (__bf16)b.x; o[5] = (__bf16)b.y; o[6] = (__bf16)b.z; o[7] = (__bf16)b.w;
  *(v8bf*)(Xb + i) = o;
}

// ---------------- fused gate+up GEMM + SiLU, per-expert grouped ----------------
// grid: (INTER/64, ceil(T/64)=32, EXPERTS), block: 128 = 4 waves.
// Wave tile: 64(M) x 16(N), gate+up: 8 accumulators; B frags reused 4x.
// K-loop is branch-free: all 4 M-subtiles computed unconditionally (rows clamped,
// garbage masked at store) so accumulators stay pinned and WMMAs are back-to-back.
__global__ __launch_bounds__(128) void moe_gateup(
    const __bf16* __restrict__ Xb, const float* __restrict__ Wgp, const float* __restrict__ Wup,
    const int* __restrict__ cnt, const int* __restrict__ rbase,
    const int* __restrict__ tok_of_row, __bf16* __restrict__ Hbuf) {
  const int e = blockIdx.z;
  const int count = cnt[e];
  const int mt = blockIdx.y;            // 64-row strip
  if (mt * 64 >= count) return;
  const int base = rbase[e];
  const int wave = threadIdx.x >> 5;
  const int l = threadIdx.x & 31;
  const int kh = l >> 4;                // K-half select
  const int m = l & 15;                 // A row within 16-row subtile
  const int n_col = blockIdx.x * 64 + wave * 16 + (l & 15);  // B/C column

  // row-gather base pointers for the 4 M-subtiles (clamped; masked at store)
  const int m0 = mt * 64 + m;
  const int r0 = base + (m0      < count ? m0      : count - 1);
  const int r1 = base + (m0 + 16 < count ? m0 + 16 : count - 1);
  const int r2 = base + (m0 + 32 < count ? m0 + 32 : count - 1);
  const int r3 = base + (m0 + 48 < count ? m0 + 48 : count - 1);
  const __bf16* a0 = Xb + (size_t)tok_of_row[r0] * HIDDEN + kh * 8;
  const __bf16* a1 = Xb + (size_t)tok_of_row[r1] * HIDDEN + kh * 8;
  const __bf16* a2 = Xb + (size_t)tok_of_row[r2] * HIDDEN + kh * 8;
  const __bf16* a3 = Xb + (size_t)tok_of_row[r3] * HIDDEN + kh * 8;
  const float* gp = Wgp + (size_t)e * HIDDEN * INTER + n_col + (size_t)(kh * 8) * INTER;
  const float* up = Wup + (size_t)e * HIDDEN * INTER + n_col + (size_t)(kh * 8) * INTER;

  v8f cg0 = {}, cg1 = {}, cg2 = {}, cg3 = {};
  v8f cu0 = {}, cu1 = {}, cu2 = {}, cu3 = {};
  for (int kk = 0; kk < HIDDEN; kk += 32) {
    // B fragments: VGPR j/2 packs k = 8*kh + j + (j & 8); stream f32 weights, cvt in-regs
    const float* gpk = gp + (size_t)kk * INTER;
    const float* upk = up + (size_t)kk * INTER;
    v16bf bg, bu;
#pragma unroll
    for (int j = 0; j < 16; ++j) {
      bg[j] = (__bf16)gpk[(size_t)(j + (j & 8)) * INTER];
      bu[j] = (__bf16)upk[(size_t)(j + (j & 8)) * INTER];
    }
    FragA f0, f1, f2, f3;
    f0.h[0] = *(const v8bf*)(a0 + kk);  f0.h[1] = *(const v8bf*)(a0 + kk + 16);
    f1.h[0] = *(const v8bf*)(a1 + kk);  f1.h[1] = *(const v8bf*)(a1 + kk + 16);
    f2.h[0] = *(const v8bf*)(a2 + kk);  f2.h[1] = *(const v8bf*)(a2 + kk + 16);
    f3.h[0] = *(const v8bf*)(a3 + kk);  f3.h[1] = *(const v8bf*)(a3 + kk + 16);
    cg0 = __builtin_amdgcn_wmma_f32_16x16x32_bf16(false, f0.v, false, bg, (short)0, cg0, false, false);
    cu0 = __builtin_amdgcn_wmma_f32_16x16x32_bf16(false, f0.v, false, bu, (short)0, cu0, false, false);
    cg1 = __builtin_amdgcn_wmma_f32_16x16x32_bf16(false, f1.v, false, bg, (short)0, cg1, false, false);
    cu1 = __builtin_amdgcn_wmma_f32_16x16x32_bf16(false, f1.v, false, bu, (short)0, cu1, false, false);
    cg2 = __builtin_amdgcn_wmma_f32_16x16x32_bf16(false, f2.v, false, bg, (short)0, cg2, false, false);
    cu2 = __builtin_amdgcn_wmma_f32_16x16x32_bf16(false, f2.v, false, bu, (short)0, cu2, false, false);
    cg3 = __builtin_amdgcn_wmma_f32_16x16x32_bf16(false, f3.v, false, bg, (short)0, cg3, false, false);
    cu3 = __builtin_amdgcn_wmma_f32_16x16x32_bf16(false, f3.v, false, bu, (short)0, cu3, false, false);
  }
  // epilogue: C layout lane -> (n = l&15, m = r + 8*kh); h = silu(g)*u
  const v8f* cgs[4] = { &cg0, &cg1, &cg2, &cg3 };
  const v8f* cus[4] = { &cu0, &cu1, &cu2, &cu3 };
#pragma unroll
  for (int s = 0; s < 4; ++s) {
#pragma unroll
    for (int r = 0; r < 8; ++r) {
      const int gmr = mt * 64 + s * 16 + r + 8 * kh;
      if (gmr < count) {
        const float g = (*cgs[s])[r], u = (*cus[s])[r];
        const float h = g * fast_sigmoid(g) * u;
        Hbuf[(size_t)(base + gmr) * INTER + n_col] = (__bf16)h;
      }
    }
  }
}

// ---------------- down GEMM, scaled by routing weight ----------------
// grid: (HIDDEN/64, 32, EXPERTS), block: 128. Wave tile: 64(M) x 16(N). Branch-free K loop.
__global__ __launch_bounds__(128) void moe_down(
    const __bf16* __restrict__ Hbuf, const float* __restrict__ Wdp,
    const int* __restrict__ cnt, const int* __restrict__ rbase,
    const float* __restrict__ w_of_row, float* __restrict__ Y) {
  const int e = blockIdx.z;
  const int count = cnt[e];
  const int mt = blockIdx.y;
  if (mt * 64 >= count) return;
  const int base = rbase[e];
  const int wave = threadIdx.x >> 5;
  const int l = threadIdx.x & 31;
  const int kh = l >> 4;
  const int m = l & 15;
  const int n_col = blockIdx.x * 64 + wave * 16 + (l & 15);

  const int m0 = mt * 64 + m;
  const int r0 = base + (m0      < count ? m0      : count - 1);
  const int r1 = base + (m0 + 16 < count ? m0 + 16 : count - 1);
  const int r2 = base + (m0 + 32 < count ? m0 + 32 : count - 1);
  const int r3 = base + (m0 + 48 < count ? m0 + 48 : count - 1);
  const __bf16* a0 = Hbuf + (size_t)r0 * INTER + kh * 8;
  const __bf16* a1 = Hbuf + (size_t)r1 * INTER + kh * 8;
  const __bf16* a2 = Hbuf + (size_t)r2 * INTER + kh * 8;
  const __bf16* a3 = Hbuf + (size_t)r3 * INTER + kh * 8;
  const float* bpb = Wdp + (size_t)e * INTER * HIDDEN + n_col + (size_t)(kh * 8) * HIDDEN;

  v8f c0 = {}, c1 = {}, c2 = {}, c3 = {};
  for (int kk = 0; kk < INTER; kk += 32) {
    const float* bp = bpb + (size_t)kk * HIDDEN;
    v16bf b;
#pragma unroll
    for (int j = 0; j < 16; ++j) b[j] = (__bf16)bp[(size_t)(j + (j & 8)) * HIDDEN];
    FragA f0, f1, f2, f3;
    f0.h[0] = *(const v8bf*)(a0 + kk);  f0.h[1] = *(const v8bf*)(a0 + kk + 16);
    f1.h[0] = *(const v8bf*)(a1 + kk);  f1.h[1] = *(const v8bf*)(a1 + kk + 16);
    f2.h[0] = *(const v8bf*)(a2 + kk);  f2.h[1] = *(const v8bf*)(a2 + kk + 16);
    f3.h[0] = *(const v8bf*)(a3 + kk);  f3.h[1] = *(const v8bf*)(a3 + kk + 16);
    c0 = __builtin_amdgcn_wmma_f32_16x16x32_bf16(false, f0.v, false, b, (short)0, c0, false, false);
    c1 = __builtin_amdgcn_wmma_f32_16x16x32_bf16(false, f1.v, false, b, (short)0, c1, false, false);
    c2 = __builtin_amdgcn_wmma_f32_16x16x32_bf16(false, f2.v, false, b, (short)0, c2, false, false);
    c3 = __builtin_amdgcn_wmma_f32_16x16x32_bf16(false, f3.v, false, b, (short)0, c3, false, false);
  }
  const v8f* cs[4] = { &c0, &c1, &c2, &c3 };
#pragma unroll
  for (int s = 0; s < 4; ++s) {
#pragma unroll
    for (int r = 0; r < 8; ++r) {
      const int gmr = mt * 64 + s * 16 + r + 8 * kh;
      if (gmr < count) {
        const int grow = base + gmr;
        Y[(size_t)grow * HIDDEN + n_col] = (*cs[s])[r] * w_of_row[grow];
      }
    }
  }
}

// ---------------- combine: out[t] = Y[slot0(t)] + Y[slot1(t)] ----------------
__global__ void moe_combine(const float* __restrict__ Y, const int* __restrict__ slot_map,
                            float* __restrict__ out) {
  const int idx = blockIdx.x * blockDim.x + threadIdx.x;  // over T * H/4
  const int t = idx / (HIDDEN / 4);
  const int h4 = idx % (HIDDEN / 4);
  const int s0 = slot_map[2 * t], s1 = slot_map[2 * t + 1];
  const float4 a = *(const float4*)(Y + (size_t)s0 * HIDDEN + h4 * 4);
  const float4 b = *(const float4*)(Y + (size_t)s1 * HIDDEN + h4 * 4);
  float4 o; o.x = a.x + b.x; o.y = a.y + b.y; o.z = a.z + b.z; o.w = a.w + b.w;
  *(float4*)(out + (size_t)t * HIDDEN + h4 * 4) = o;
}

extern "C" void kernel_launch(void* const* d_in, const int* in_sizes, int n_in,
                              void* d_out, int out_size, void* d_ws, size_t ws_size,
                              hipStream_t stream) {
  (void)in_sizes; (void)n_in; (void)out_size; (void)ws_size;
  const float* X   = (const float*)d_in[0];   // [T, H]
  const float* Wg  = (const float*)d_in[1];   // [H, E]
  const float* Wgp = (const float*)d_in[2];   // [E, H, I]
  const float* Wup = (const float*)d_in[3];   // [E, H, I]
  const float* Wdp = (const float*)d_in[4];   // [E, I, H]
  float* out = (float*)d_out;                 // [T, H]

  char* ws = (char*)d_ws;
  size_t off = 0;
  auto alloc = [&](size_t bytes) -> char* {
    off = (off + 255) & ~(size_t)255;
    char* p = ws + off;
    off += bytes;
    return p;
  };
  int*    cnt        = (int*)alloc(EXPERTS * sizeof(int));
  int*    cnt2       = (int*)alloc(EXPERTS * sizeof(int));
  int*    rbase      = (int*)alloc((EXPERTS + 1) * sizeof(int));
  int*    tk_id      = (int*)alloc(NROWS * sizeof(int));
  float*  tk_w       = (float*)alloc(NROWS * sizeof(float));
  int*    tok_of_row = (int*)alloc(NROWS * sizeof(int));
  float*  w_of_row   = (float*)alloc(NROWS * sizeof(float));
  int*    slot_map   = (int*)alloc(NROWS * sizeof(int));
  __bf16* Xb         = (__bf16*)alloc((size_t)TOKENS * HIDDEN * sizeof(__bf16));
  __bf16* Hbuf       = (__bf16*)alloc((size_t)NROWS * INTER * sizeof(__bf16));
  float*  Y          = (float*)alloc((size_t)NROWS * HIDDEN * sizeof(float));

  // 1) router
  moe_router<<<TOKENS, 32, 0, stream>>>(X, Wg, tk_id, tk_w);
  // 2) routing metadata
  moe_zero_meta<<<1, 32, 0, stream>>>(cnt, cnt2);
  moe_count<<<TOKENS / 256, 256, 0, stream>>>(tk_id, cnt);
  moe_prefix<<<1, 32, 0, stream>>>(cnt, rbase);
  moe_scatter<<<TOKENS / 256, 256, 0, stream>>>(tk_id, tk_w, rbase, cnt2,
                                                tok_of_row, w_of_row, slot_map);
  // 3) activations to bf16
  moe_cvt_x<<<(TOKENS * HIDDEN / 8) / 256, 256, 0, stream>>>(X, Xb);
  // 4) fused gate/up + SiLU (grid.y = ceil(T/64); early-exit; expert slowest dim for L2 reuse)
  moe_gateup<<<dim3(INTER / 64, 32, EXPERTS), 128, 0, stream>>>(
      Xb, Wgp, Wup, cnt, rbase, tok_of_row, Hbuf);
  // 5) down projection, scaled
  moe_down<<<dim3(HIDDEN / 64, 32, EXPERTS), 128, 0, stream>>>(
      Hbuf, Wdp, cnt, rbase, w_of_row, Y);
  // 6) combine two expert slots per token
  moe_combine<<<(TOKENS * (HIDDEN / 4)) / 256, 256, 0, stream>>>(Y, slot_map, out);
}
